// arDCA_74062416053356
// MI455X (gfx1250) — compile-verified, matrix operand order checked by reading, not run on hardware
//
#include <hip/hip_runtime.h>
#include <math.h>

typedef __attribute__((ext_vector_type(16))) _Float16 v16h;
typedef __attribute__((ext_vector_type(8)))  float    v8f;

// ---- problem constants (from reference) ----
constexpr int N_SEQ = 4096;   // sequences
constexpr int RES   = 400;    // known residues (= i)
constexpr int Q     = 21;     // alphabet
constexpr int L     = 512;    // total residues
constexpr int K     = RES * Q;            // 8400 reduction length
constexpr int KT    = (K + 31) / 32;      // 263 k-tiles of 32
constexpr int NCHUNK = 4;                 // K split for parallelism
constexpr int CH    = (KT + NCHUNK - 1) / NCHUNK;   // 66 k-tiles per chunk
constexpr int MT    = N_SEQ / 16;         // 256 m-tiles
constexpr int WAVES = 4;                  // waves per block in gemm

// ---- workspace layout ----
// [0, BP_BYTES):            packed f16 B fragments: [kt][ntile(2)][lane(32)][16 halves]
// [PART_OFF, +PART_BYTES):  f32 partial logits: [chunk(4)][n(4096)][32]
constexpr size_t BP_BYTES   = (size_t)KT * 2 * 32 * 16 * 2;        // 538,624 B
constexpr size_t PART_OFF   = BP_BYTES;                            // 256B-aligned already
constexpr size_t PART_BYTES = (size_t)NCHUNK * N_SEQ * 32 * 4;     // 2 MB

// -------------------------------------------------------------------------
// Kernel 1: pack B = J[i, :, :i, :]^T (K x 21, padded to K'=8416 x 32) into
// per-lane WMMA B fragments matching the 16-bit operand VGPR layout:
//   lane n in [0,15] holds column n; group g = lane>>4 selects K halves;
//   element e in [0,7]  -> k_local = g*8 + e
//   element e in [8,15] -> k_local = 16 + g*8 + (e-8)
// -------------------------------------------------------------------------
__global__ void pack_B_kernel(const float* __restrict__ J, _Float16* __restrict__ Bp) {
    int idx = blockIdx.x * blockDim.x + threadIdx.x;
    if (idx >= KT * 2 * 32) return;
    const int kt    = idx >> 6;
    const int ntile = (idx >> 5) & 1;
    const int lane  = idx & 31;
    const int n     = (lane & 15) + ntile * 16;   // output class (column of B)
    const int g     = lane >> 4;

    v16h frag;
#pragma unroll
    for (int e = 0; e < 16; ++e) {
        const int klocal = (e < 8) ? (g * 8 + e) : (16 + g * 8 + (e - 8));
        const int k = kt * 32 + klocal;
        float x = 0.0f;
        if (n < Q && k < K) {
            const int j = k / Q;
            const int b = k - j * Q;
            // J flat index: ((l1*Q + q1)*L + l2)*Q + q2, with l1=RES, q1=n, l2=j, q2=b
            x = J[((size_t)(RES * Q + n) * L + (size_t)j) * Q + b];
        }
        frag[e] = (_Float16)x;
    }
    *reinterpret_cast<v16h*>(Bp + (size_t)idx * 16) = frag;
}

// -------------------------------------------------------------------------
// Helpers for the GEMM inner step
// -------------------------------------------------------------------------
__device__ __forceinline__ v16h load_a_full(const float* __restrict__ xrow, int base) {
    const float4 r1a = *reinterpret_cast<const float4*>(xrow + base);
    const float4 r1b = *reinterpret_cast<const float4*>(xrow + base + 4);
    const float4 r2a = *reinterpret_cast<const float4*>(xrow + base + 16);
    const float4 r2b = *reinterpret_cast<const float4*>(xrow + base + 20);
    v16h a;
    a[0]  = (_Float16)r1a.x; a[1]  = (_Float16)r1a.y;
    a[2]  = (_Float16)r1a.z; a[3]  = (_Float16)r1a.w;
    a[4]  = (_Float16)r1b.x; a[5]  = (_Float16)r1b.y;
    a[6]  = (_Float16)r1b.z; a[7]  = (_Float16)r1b.w;
    a[8]  = (_Float16)r2a.x; a[9]  = (_Float16)r2a.y;
    a[10] = (_Float16)r2a.z; a[11] = (_Float16)r2a.w;
    a[12] = (_Float16)r2b.x; a[13] = (_Float16)r2b.y;
    a[14] = (_Float16)r2b.z; a[15] = (_Float16)r2b.w;
    return a;
}

// -------------------------------------------------------------------------
// Kernel 2: WMMA GEMM. One wave per 16-row M-tile per K-chunk.
// A fragment streamed from f32 X (exact one-hot -> f16), B fragments from Bp.
// Two accumulators cover output columns 0..15 and 16..31 (>=21 are zero).
// Tail k-tile (zero-padded K) is peeled so the hot loop is branch-free.
// -------------------------------------------------------------------------
__global__ void __launch_bounds__(WAVES * 32)
gemm_kernel(const float* __restrict__ X, const _Float16* __restrict__ Bp,
            float* __restrict__ partial) {
    const int lane  = threadIdx.x & 31;
    const int wave  = threadIdx.x >> 5;
    const int mtile = blockIdx.x * WAVES + wave;
    const int chunk = blockIdx.y;
    const int g     = lane >> 4;
    const int mrow  = mtile * 16 + (lane & 15);
    const float* __restrict__ xrow = X + (size_t)mrow * K;

    const int kt0      = chunk * CH;
    const int kt1      = (kt0 + CH < KT) ? (kt0 + CH) : KT;   // chunk end
    const int kt_main  = (kt1 < KT) ? kt1 : (KT - 1);         // branch-free portion

    v8f c0 = {};
    v8f c1 = {};

#pragma unroll 2
    for (int kt = kt0; kt < kt_main; ++kt) {
        const int base = kt * 32 + g * 8;
        // stream prefetch ~2KB ahead on the X row (speculative; OOB is dropped)
        __builtin_prefetch(xrow + base + 512, 0, 1);

        const v16h a  = load_a_full(xrow, base);
        const v16h b0 = *reinterpret_cast<const v16h*>(Bp + ((size_t)(kt * 2 + 0) * 32 + lane) * 16);
        const v16h b1 = *reinterpret_cast<const v16h*>(Bp + ((size_t)(kt * 2 + 1) * 32 + lane) * 16);

        c0 = __builtin_amdgcn_wmma_f32_16x16x32_f16(false, a, false, b0, (short)0, c0, false, false);
        c1 = __builtin_amdgcn_wmma_f32_16x16x32_f16(false, a, false, b1, (short)0, c1, false, false);
    }

    if (kt1 == KT) {
        // peeled tail tile kt = KT-1: k in [8384, 8416); only elements 0..7
        // (k = 8384 + g*8 .. +7 <= 8399) are in-range, elements 8..15 are pad.
        const int kt   = KT - 1;
        const int base = kt * 32 + g * 8;
        const float4 r1a = *reinterpret_cast<const float4*>(xrow + base);
        const float4 r1b = *reinterpret_cast<const float4*>(xrow + base + 4);
        v16h a;
        a[0] = (_Float16)r1a.x; a[1] = (_Float16)r1a.y;
        a[2] = (_Float16)r1a.z; a[3] = (_Float16)r1a.w;
        a[4] = (_Float16)r1b.x; a[5] = (_Float16)r1b.y;
        a[6] = (_Float16)r1b.z; a[7] = (_Float16)r1b.w;
#pragma unroll
        for (int e = 8; e < 16; ++e) a[e] = (_Float16)0.0f;

        const v16h b0 = *reinterpret_cast<const v16h*>(Bp + ((size_t)(kt * 2 + 0) * 32 + lane) * 16);
        const v16h b1 = *reinterpret_cast<const v16h*>(Bp + ((size_t)(kt * 2 + 1) * 32 + lane) * 16);

        c0 = __builtin_amdgcn_wmma_f32_16x16x32_f16(false, a, false, b0, (short)0, c0, false, false);
        c1 = __builtin_amdgcn_wmma_f32_16x16x32_f16(false, a, false, b1, (short)0, c1, false, false);
    }

    // C/D layout: element r -> M = g*8 + r ; column = (lane&15) (+16 for c1)
    float* prow = partial
        + ((size_t)chunk * N_SEQ + (size_t)mtile * 16 + (size_t)g * 8) * 32
        + (lane & 15);
#pragma unroll
    for (int r = 0; r < 8; ++r) {
        prow[(size_t)r * 32]      = c0[r];
        prow[(size_t)r * 32 + 16] = c1[r];
    }
}

// -------------------------------------------------------------------------
// Kernel 3: reduce chunk partials, add h[RES], softmax over 21 classes.
// -------------------------------------------------------------------------
__global__ void softmax_kernel(const float* __restrict__ partial,
                               const float* __restrict__ h,
                               float* __restrict__ out) {
    const int n = blockIdx.x * blockDim.x + threadIdx.x;
    if (n >= N_SEQ) return;

    float logit[Q];
#pragma unroll
    for (int a = 0; a < Q; ++a) logit[a] = h[RES * Q + a];

#pragma unroll
    for (int c = 0; c < NCHUNK; ++c) {
        const float* p = partial + ((size_t)c * N_SEQ + n) * 32;
#pragma unroll
        for (int a = 0; a < Q; ++a) logit[a] += p[a];
    }

    float mx = logit[0];
#pragma unroll
    for (int a = 1; a < Q; ++a) mx = fmaxf(mx, logit[a]);

    float e[Q];
    float s = 0.0f;
#pragma unroll
    for (int a = 0; a < Q; ++a) {          // BETA == 1.0
        e[a] = __expf(logit[a] - mx);
        s += e[a];
    }
    const float inv = 1.0f / s;
#pragma unroll
    for (int a = 0; a < Q; ++a) out[(size_t)n * Q + a] = e[a] * inv;
}

// -------------------------------------------------------------------------
extern "C" void kernel_launch(void* const* d_in, const int* in_sizes, int n_in,
                              void* d_out, int out_size, void* d_ws, size_t ws_size,
                              hipStream_t stream) {
    (void)in_sizes; (void)n_in; (void)out_size; (void)ws_size; (void)PART_BYTES;

    const float* X = (const float*)d_in[0];   // [4096, 400, 21] f32 one-hot
    const float* h = (const float*)d_in[1];   // [512, 21] f32
    const float* J = (const float*)d_in[2];   // [512, 21, 512, 21] f32
    float* out = (float*)d_out;               // [4096, 21] f32

    _Float16* Bp   = (_Float16*)d_ws;
    float*    part = (float*)((char*)d_ws + PART_OFF);

    {   // pack B fragments (tiny)
        const int total = KT * 2 * 32;
        const int thr = 256;
        pack_B_kernel<<<(total + thr - 1) / thr, thr, 0, stream>>>(J, Bp);
    }
    {   // WMMA GEMM: 64 blocks x 4 chunks, 4 waves/block -> 1024 waves
        dim3 grid(MT / WAVES, NCHUNK);
        gemm_kernel<<<grid, WAVES * 32, 0, stream>>>(X, Bp, part);
    }
    {   // reduce + softmax
        softmax_kernel<<<N_SEQ / 256, 256, 0, stream>>>(part, h, out);
    }
}
